// SplineNet_69045894250551
// MI455X (gfx1250) — matrix operand order repaired and checked by reading, not compiled, not run on hardware
//
#include <hip/hip_runtime.h>

typedef __bf16 bf16_t;
typedef __attribute__((ext_vector_type(16))) __bf16 v16bf;
typedef __attribute__((ext_vector_type(8)))  __bf16 v8bf;
typedef __attribute__((ext_vector_type(8)))  float  v8f;

#define N_NODES 50000
#define N_EDGES 800000

// ---------------- utility kernels ----------------

__global__ void zero_f32(float* __restrict__ p, int n) {
  int i = blockIdx.x * blockDim.x + threadIdx.x;
  if (i < n) p[i] = 0.f;
}

__global__ void cvt_f32_bf16(const float* __restrict__ in, bf16_t* __restrict__ out, int n) {
  int i = blockIdx.x * blockDim.x + threadIdx.x;
  if (i < n) out[i] = (bf16_t)in[i];
}

// Transpose + convert weights: Wt[s][dout*64+din] = W[s][din*64+dout].
// Slice KK (the last one) comes from `root`.
__global__ void prep_weights(const float* __restrict__ W, const float* __restrict__ root,
                             bf16_t* __restrict__ Wt, int KK) {
  int gid = blockIdx.x * blockDim.x + threadIdx.x;
  int total = (KK + 1) << 12;
  if (gid >= total) return;
  int s    = gid >> 12;
  int r    = gid & 4095;
  int dout = r >> 6;
  int din  = r & 63;
  float v = (s < KK) ? W[(s << 12) + (din << 6) + dout] : root[(din << 6) + dout];
  Wt[gid] = (bf16_t)v;
}

// ---------------- WMMA GEMM: Y[m, s, 0:64] = A[m,:] @ W_s ----------------
// A: [nrows, 64] bf16 row-major.  Wt: [S][64,64] bf16, stored TRANSPOSED
// (row = dout, contiguous din) so B-tile lane loads are contiguous.
// Y: [nrows, S, 64] bf16.

// 16-bit A 16x32 layout: lane<16 -> K {kb*32+0..7, +16..23}; lane>=16 -> {+8..15, +24..31}
__device__ __forceinline__ v16bf load_a_tile(const bf16_t* __restrict__ A,
                                             int r0, int kb, int lane) {
  const bf16_t* p = A + ((r0 + (lane & 15)) << 6) + (kb << 5) + ((lane >> 4) << 3);
  v8bf lo = *(const v8bf*)p;
  v8bf hi = *(const v8bf*)(p + 16);
  v16bf t;
#pragma unroll
  for (int i = 0; i < 8; ++i) { t[i] = lo[i]; t[i + 8] = hi[i]; }
  return t;
}

// 16-bit B 32x16 layout: lane = column (N), lanes 0-15 hold K=kb*32+{0..7,16..23},
// lanes 16-31 hold K=kb*32+{8..15,24..31} -> two 8-bf16 runs in transposed storage.
__device__ __forceinline__ v16bf load_b_tile(const bf16_t* __restrict__ Bt,
                                             int nt, int kb, int lane) {
  const bf16_t* p = Bt + (((nt << 4) + (lane & 15)) << 6) + (kb << 5) + ((lane >> 4) << 3);
  v8bf lo = *(const v8bf*)p;
  v8bf hi = *(const v8bf*)(p + 16);
  v16bf t;
#pragma unroll
  for (int i = 0; i < 8; ++i) { t[i] = lo[i]; t[i + 8] = hi[i]; }
  return t;
}

__global__ __launch_bounds__(256)
void gemm_x_w(const bf16_t* __restrict__ A, const bf16_t* __restrict__ Wt,
              bf16_t* __restrict__ Y, int nrows, int S) {
  const int lane = threadIdx.x & 31;
  const int wid  = threadIdx.x >> 5;
  const int r0   = ((blockIdx.x << 3) + wid) << 4;   // 16-row tile per wave
  if (r0 >= nrows) return;                           // wave-uniform; EXEC stays all-1s
  const int s = blockIdx.y;
  const bf16_t* Bt = Wt + ((size_t)s << 12);

  v16bf a0 = load_a_tile(A, r0, 0, lane);
  v16bf a1 = load_a_tile(A, r0, 1, lane);

#pragma unroll
  for (int nt = 0; nt < 4; ++nt) {
    v8f acc = {};
    v16bf b0 = load_b_tile(Bt, nt, 0, lane);
    acc = __builtin_amdgcn_wmma_f32_16x16x32_bf16(false, a0, false, b0,
                                                  (short)0, acc, false, false);
    v16bf b1 = load_b_tile(Bt, nt, 1, lane);
    acc = __builtin_amdgcn_wmma_f32_16x16x32_bf16(false, a1, false, b1,
                                                  (short)0, acc, false, false);
    // C layout: VGPR r -> M = r (lanes 0-15) / r+8 (lanes 16-31); N = lane&15
    const int n  = (nt << 4) + (lane & 15);
    const int mb = r0 + ((lane >> 4) << 3);
#pragma unroll
    for (int r = 0; r < 8; ++r)
      Y[((size_t)(mb + r) * S + s) * 64 + n] = (bf16_t)acc[r];
  }
}

// ---------------- per-edge spline gather + scatter-mean ----------------
// One wave (32 lanes) per edge; each lane handles 2 of the 64 channels.
__global__ void edge_aggregate(const int* __restrict__ ei, const float* __restrict__ ea,
                               const bf16_t* __restrict__ Y, int S, int K,
                               float* __restrict__ msum, float* __restrict__ cnt) {
  long long gid = (long long)blockIdx.x * blockDim.x + threadIdx.x;
  int e    = (int)(gid >> 5);
  int lane = (int)(gid & 31);
  if (e >= N_EDGES) return;
  int src = ei[e];
  int dst = ei[N_EDGES + e];

  // degree-1 open B-spline basis, E_DIM = 2
  float Km1 = (float)(K - 1);
  float v0 = ea[2 * e] * Km1, v1 = ea[2 * e + 1] * Km1;
  float f0 = floorf(v0), f1 = floorf(v1);
  float fr0 = v0 - f0, fr1 = v1 - f1;
  int l0 = (int)f0, l1 = (int)f1;
  int i00 = min(max(l0, 0), K - 1), i01 = min(max(l0 + 1, 0), K - 1);
  int i10 = min(max(l1, 0), K - 1), i11 = min(max(l1 + 1, 0), K - 1);
  float B0 = (1.f - fr0) * (1.f - fr1);
  float B1 = (1.f - fr0) * fr1;
  float B2 = fr0 * (1.f - fr1);
  float B3 = fr0 * fr1;
  int WI0 = i00 + K * i10, WI1 = i00 + K * i11;
  int WI2 = i01 + K * i10, WI3 = i01 + K * i11;

  int c = lane << 1;
  const bf16_t* Yb = Y + ((size_t)src * S << 6) + c;
  float a0 = 0.f, a1 = 0.f;
  const bf16_t* p;
  p = Yb + ((size_t)WI0 << 6); a0 = fmaf(B0, (float)p[0], a0); a1 = fmaf(B0, (float)p[1], a1);
  p = Yb + ((size_t)WI1 << 6); a0 = fmaf(B1, (float)p[0], a0); a1 = fmaf(B1, (float)p[1], a1);
  p = Yb + ((size_t)WI2 << 6); a0 = fmaf(B2, (float)p[0], a0); a1 = fmaf(B2, (float)p[1], a1);
  p = Yb + ((size_t)WI3 << 6); a0 = fmaf(B3, (float)p[0], a0); a1 = fmaf(B3, (float)p[1], a1);

  atomicAdd(&msum[((size_t)dst << 6) + c], a0);
  atomicAdd(&msum[((size_t)dst << 6) + c + 1], a1);
  if (lane == 0) atomicAdd(&cnt[dst], 1.0f);
}

// mean + root-term (last Y slice) + bias, then ELU -> bf16 for next GEMM
__global__ void combine_elu(const float* __restrict__ msum, const float* __restrict__ cnt,
                            const bf16_t* __restrict__ Y, int S,
                            const float* __restrict__ bias, bf16_t* __restrict__ h) {
  int gid = blockIdx.x * blockDim.x + threadIdx.x;
  if (gid >= N_NODES * 64) return;
  int n = gid >> 6, d = gid & 63;
  float cv  = fmaxf(cnt[n], 1.0f);
  float rt  = (float)Y[((size_t)n * S + (S - 1)) * 64 + d];
  float val = msum[gid] / cv + rt + bias[d];
  val = (val > 0.f) ? val : (expf(val) - 1.0f);
  h[gid] = (bf16_t)val;
}

__global__ void bias_relu(const bf16_t* __restrict__ Y, const float* __restrict__ bias,
                          bf16_t* __restrict__ h) {
  int gid = blockIdx.x * blockDim.x + threadIdx.x;
  if (gid >= N_NODES * 64) return;
  int d = gid & 63;
  float val = fmaxf((float)Y[gid] + bias[d], 0.f);
  h[gid] = (bf16_t)val;
}

// final tiny projection: out[n, 0:8] = relu(h @ mlp2_w + b)
__global__ void mlp2_out(const bf16_t* __restrict__ h, const float* __restrict__ w,
                         const float* __restrict__ b, float* __restrict__ out) {
  int gid = blockIdx.x * blockDim.x + threadIdx.x;
  if (gid >= N_NODES * 8) return;
  int n = gid >> 3, cc = gid & 7;
  float acc = b[cc];
  const bf16_t* row = h + ((size_t)n << 6);
#pragma unroll
  for (int d = 0; d < 64; ++d) acc = fmaf((float)row[d], w[d * 8 + cc], acc);
  out[gid] = fmaxf(acc, 0.f);
}

// ---------------- host orchestration ----------------
extern "C" void kernel_launch(void* const* d_in, const int* in_sizes, int n_in,
                              void* d_out, int out_size, void* d_ws, size_t ws_size,
                              hipStream_t stream) {
  (void)in_sizes; (void)n_in; (void)out_size; (void)ws_size;
  const float* x      = (const float*)d_in[0];
  const int*   ei     = (const int*)  d_in[1];
  const float* ea     = (const float*)d_in[2];
  const float* W1     = (const float*)d_in[3];
  const float* root1  = (const float*)d_in[4];
  const float* b1     = (const float*)d_in[5];
  const float* W2     = (const float*)d_in[6];
  const float* root2  = (const float*)d_in[7];
  const float* b2     = (const float*)d_in[8];
  const float* mlp1w  = (const float*)d_in[9];
  const float* mlp1b  = (const float*)d_in[10];
  const float* mlp2w  = (const float*)d_in[11];
  const float* mlp2b  = (const float*)d_in[12];
  float* out = (float*)d_out;

  char* ws = (char*)d_ws;
  size_t off = 0;
  auto alloc = [&](size_t bytes) -> void* {
    void* p = ws + off;
    off += (bytes + 255) & ~(size_t)255;
    return p;
  };
  bf16_t* x_bf = (bf16_t*)alloc((size_t)N_NODES * 64 * 2);
  bf16_t* h_bf = (bf16_t*)alloc((size_t)N_NODES * 64 * 2);
  bf16_t* Wt   = (bf16_t*)alloc((size_t)26 * 4096 * 2);
  bf16_t* Y    = (bf16_t*)alloc((size_t)N_NODES * 26 * 64 * 2);  // ~166 MB, L2-resident
  float*  msum = (float*) alloc((size_t)N_NODES * 64 * 4);
  float*  cnt  = (float*) alloc((size_t)N_NODES * 4);

  const int TB = 256;
  const int gND = (N_NODES * 64 + TB - 1) / TB;     // node*feat grid
  const int gGemm = (3125 + 7) / 8;                 // 50000/16 row tiles, 8 waves/block
  const int gEdge = (N_EDGES * 32) / TB;            // one wave per edge

  cvt_f32_bf16<<<gND, TB, 0, stream>>>(x, x_bf, N_NODES * 64);

  // ---- layer 1: K=3, 9 spline kernels + root as slice 9 ----
  prep_weights<<<(10 * 4096 + TB - 1) / TB, TB, 0, stream>>>(W1, root1, Wt, 9);
  gemm_x_w<<<dim3(gGemm, 10), TB, 0, stream>>>(x_bf, Wt, Y, N_NODES, 10);
  zero_f32<<<gND, TB, 0, stream>>>(msum, N_NODES * 64);
  zero_f32<<<(N_NODES + TB - 1) / TB, TB, 0, stream>>>(cnt, N_NODES);
  edge_aggregate<<<gEdge, TB, 0, stream>>>(ei, ea, Y, 10, 3, msum, cnt);
  combine_elu<<<gND, TB, 0, stream>>>(msum, cnt, Y, 10, b1, h_bf);

  // ---- layer 2: K=5, 25 spline kernels + root as slice 25 ----
  prep_weights<<<(26 * 4096 + TB - 1) / TB, TB, 0, stream>>>(W2, root2, Wt, 25);
  gemm_x_w<<<dim3(gGemm, 26), TB, 0, stream>>>(h_bf, Wt, Y, N_NODES, 26);
  zero_f32<<<gND, TB, 0, stream>>>(msum, N_NODES * 64);
  zero_f32<<<(N_NODES + TB - 1) / TB, TB, 0, stream>>>(cnt, N_NODES);
  edge_aggregate<<<gEdge, TB, 0, stream>>>(ei, ea, Y, 26, 5, msum, cnt);
  combine_elu<<<gND, TB, 0, stream>>>(msum, cnt, Y, 26, b2, x_bf);  // x_bf now holds h2

  // ---- MLP head ----
  prep_weights<<<(4096 + TB - 1) / TB, TB, 0, stream>>>(mlp1w, mlp1w, Wt, 0); // slice 0 = mlp1w^T
  gemm_x_w<<<dim3(gGemm, 1), TB, 0, stream>>>(x_bf, Wt, Y, N_NODES, 1);
  bias_relu<<<gND, TB, 0, stream>>>(Y, mlp1b, h_bf);
  mlp2_out<<<(N_NODES * 8 + TB - 1) / TB, TB, 0, stream>>>(h_bf, mlp2w, mlp2b, out);
}
